// MemoryReader_39539468927573
// MI455X (gfx1250) — matrix operand order repaired and checked by reading, not compiled, not run on hardware
//
#include <hip/hip_runtime.h>
#include <hip/hip_bf16.h>

typedef __attribute__((ext_vector_type(16))) __bf16 v16bf;
typedef __attribute__((ext_vector_type(8)))  float  v8f;

#define N_B     2
#define N_CK    64
#define N_CV    512
#define N_HEADS 4
#define N_CH    16      // CK per head
#define N_CVH   128     // CV per head
#define N_HW    1024
#define N_THW   8192
#define N_CHUNK 256     // THW / 32
#define N_SPLIT 4       // split-K factor over the memory axis

#if __has_builtin(__builtin_amdgcn_exp2f)
#define EXP2F(x) __builtin_amdgcn_exp2f(x)
#else
#define EXP2F(x) exp2f(x)
#endif
#define MS_SCALE 0.180336880111170232f   // 0.125 * log2(e)

// Fragment K map: group g (lane half), element e in [0,16):
//   K(g,e) = 8*g + e + (e>=8 ? 8 : 0)   ->  g=0: {0..7,16..23}, g=1: {8..15,24..31}
// matches the CDNA5 16-bit A 16x32 VGPR layout (ISA 7.12.2).

// ---------------- pre-pass: mv f32 -> bf16 in A-fragment order ----------------
// pmv layout: [b*4+h][chunk][cvrow 0..127][g][16 bf16]; 32B per thread.
__global__ __launch_bounds__(256)
void pack_mv_kernel(const float* __restrict__ mv, __bf16* __restrict__ pmv)
{
    int idx = blockIdx.x * 256 + threadIdx.x;       // 524288 threads
    int g   = idx & 1;
    int row = (idx >> 1) & 127;
    int ch  = (idx >> 8) & 255;
    int bh  = idx >> 16;
    const float* src = mv + ((size_t)bh * N_CVH + row) * N_THW + ch * 32 + 8 * g;
    v16bf v;
#pragma unroll
    for (int e = 0; e < 8; ++e) v[e] = (__bf16)src[e];          // K = 8g+e
#pragma unroll
    for (int e = 8; e < 16; ++e) v[e] = (__bf16)src[e + 8];     // K = 8g+8+e
    *(v16bf*)(pmv + (size_t)idx * 16) = v;
}

// ---------------- pre-pass: [mk | -mk^3] bf16 in A-fragment order ----------------
// pmk layout: [b*4+h][chunk][n 0..31][g][16 bf16]
__global__ __launch_bounds__(256)
void pack_mk_kernel(const float* __restrict__ mk, __bf16* __restrict__ pmk)
{
    int idx = blockIdx.x * 256 + threadIdx.x;       // 131072 threads
    int g   = idx & 1;
    int n   = (idx >> 1) & 31;
    int ch  = (idx >> 6) & 255;
    int bh  = idx >> 14;
    const float* src = mk + ((size_t)bh * N_CH + 8 * g) * N_THW + ch * 32 + n;
    float x[8];
#pragma unroll
    for (int j = 0; j < 8; ++j) x[j] = src[(size_t)j * N_THW];
    v16bf v;
#pragma unroll
    for (int e = 0; e < 8; ++e) v[e] = (__bf16)x[e];                       // K<16: mk
#pragma unroll
    for (int e = 8; e < 16; ++e) { float t = x[e - 8]; v[e] = (__bf16)(-t * t * t); } // K>=16: -mk^3
    *(v16bf*)(pmk + (size_t)idx * 16) = v;
}

// ---------------- fast fused main kernel (packed operands, split-K) ----------------
__global__ __launch_bounds__(256)
void mr_main_fast(const float* __restrict__ qk, const float* __restrict__ ms,
                  const float* __restrict__ qe, const __bf16* __restrict__ pmk,
                  const __bf16* __restrict__ pmv, float* __restrict__ out)
{
    __shared__ float expls[2][N_HEADS * 32 * 16];            // double-buffered exp(sim)
    __shared__ __align__(32) __bf16 affls[2][N_HEADS * 16 * 2 * 16]; // [h][N][g][e]

    const int tid  = threadIdx.x;
    const int wave = tid >> 5, lane = tid & 31;
    const int head = wave >> 1, q = wave & 1;
    const int hi = lane >> 4, ml = lane & 15;
    const int bm = blockIdx.x & 127;                 // (b, m-tile)
    const int s  = blockIdx.x >> 7;                  // split-K segment
    const int b  = bm >> 6;
    const int m0 = (bm & 63) * 16;
    const int m  = m0 + ml;

    // prologue: B_sim fragment + b_sq for this lane's query column m
    const float* qeb = qe + ((size_t)(b * N_HEADS + head) * N_CH) * N_HW + m;
    const float* qkb = qk + ((size_t)(b * N_HEADS + head) * N_CH) * N_HW + m;
    float b_sq = 0.f;
    v16bf bsim;
#pragma unroll
    for (int c = 0; c < 16; ++c) {
        float e = qeb[c * N_HW];
        float k = qkb[c * N_HW];
        b_sq += e * k * k * k;
        if ((c >> 3) == hi) { bsim[c & 7] = (__bf16)(2.f * k * e); bsim[(c & 7) + 8] = (__bf16)e; }   // K<16: 2*qk*qe ; K>=16: qe (table order)
    }

    const float* msb = ms + (size_t)b * N_THW + q * 16 + 8 * hi;
    const __bf16* pmkb = pmk + (size_t)(b * N_HEADS + head) * (N_CHUNK * 32 * 32)
                             + ((size_t)(q * 16 + ml) * 2 + hi) * 16;
    const __bf16* pmvb = pmv + (size_t)(b * N_HEADS + head) * (N_CHUNK * N_CVH * 32)
                             + ((size_t)(q * 64 + ml) * 2 + hi) * 16;

    v8f acc[4];
#pragma unroll
    for (int t = 0; t < 4; ++t) acc[t] = v8f{};

    const int ch0 = s * (N_CHUNK / N_SPLIT);
    const int ch1 = ch0 + (N_CHUNK / N_SPLIT);

    for (int ch = ch0; ch < ch1; ++ch) {
        const int buf = ch & 1;

        // ms values for this lane's 8 C-rows: 8 consecutive floats (broadcast loads)
        float4 msa = *(const float4*)(msb + ch * 32);
        float4 msc = *(const float4*)(msb + ch * 32 + 4);
        float msr[8] = {msa.x, msa.y, msa.z, msa.w, msc.x, msc.y, msc.z, msc.w};

        // similarity WMMA: A = packed [mk|-mk^3] fragment, single 32B load
        v16bf asim = *(const v16bf*)(pmkb + (size_t)ch * 1024);
        v8f zero{};
        v8f sim = __builtin_amdgcn_wmma_f32_16x16x32_bf16(
            false, asim, false, bsim, (short)0, zero, false, false);

#pragma unroll
        for (int r = 0; r < 8; ++r) {
            float s2 = (sim[r] - b_sq) * (msr[r] * MS_SCALE);   // log2-domain
            int nl = q * 16 + r + 8 * hi;
            expls[buf][(head * 32 + nl) * 16 + ml] = EXP2F(s2);
        }
        __syncthreads();

        // cross-head softmax; write affinity straight into B-fragment order (bf16)
#pragma unroll
        for (int p = tid; p < 512; p += 256) {
            float d = expls[buf][p] + expls[buf][512 + p] +
                      expls[buf][1024 + p] + expls[buf][1536 + p];
            float rd = 1.0f / d;
            int nl = p >> 4, mlp = p & 15;
            int gg = (nl >> 3) & 1;
            int ee = (nl & 7) | ((nl & 16) >> 1);
#pragma unroll
            for (int h = 0; h < 4; ++h)
                affls[buf][((h * 16 + mlp) * 2 + gg) * 16 + ee] =
                    (__bf16)(expls[buf][h * 512 + p] * rd);
        }
        __syncthreads();

        // readout: B fragment = one 32B LDS read; load all 4 A fragments up front,
        // then issue the 4 WMMAs back-to-back.
        v16bf baff = *(const v16bf*)(&affls[buf][((head * 16 + ml) * 2 + hi) * 16]);
        const __bf16* pv = pmvb + (size_t)ch * 4096;
        if (ch + 1 < ch1) __builtin_prefetch(pv + 4096, 0, 3);
        v16bf amv[4];
#pragma unroll
        for (int t = 0; t < 4; ++t) amv[t] = *(const v16bf*)(pv + t * 512);
#pragma unroll
        for (int t = 0; t < 4; ++t)
            acc[t] = __builtin_amdgcn_wmma_f32_16x16x32_bf16(
                false, amv[t], false, baff, (short)0, acc[t], false, false);
        // no trailing barrier: next iteration uses the other buffer
    }

    // split-K combine: atomic accumulate into zero-initialized output
    float* ob = out + ((size_t)b * 2 * N_CV + head * N_CVH + q * 64) * N_HW + m0;
#pragma unroll
    for (int t = 0; t < 4; ++t)
#pragma unroll
        for (int r = 0; r < 8; ++r)
            atomicAdd(&ob[(size_t)(t * 16 + r + 8 * hi) * N_HW + ml], acc[t][r]);
}

// ---------------- fallback fused kernel (no workspace needed) ----------------
__global__ __launch_bounds__(256)
void mr_main_kernel(const float* __restrict__ mk, const float* __restrict__ qk,
                    const float* __restrict__ ms, const float* __restrict__ qe,
                    const float* __restrict__ mv, float* __restrict__ out)
{
    __shared__ float expls[N_HEADS * 32 * 16];
    __shared__ float rdls[32 * 16];

    const int tid  = threadIdx.x;
    const int wave = tid >> 5, lane = tid & 31;
    const int head = wave >> 1, q = wave & 1;
    const int hi = lane >> 4, ml = lane & 15;
    const int b  = blockIdx.x >> 6;
    const int m0 = (blockIdx.x & 63) * 16;
    const int m  = m0 + ml;

    const float* qeb = qe + ((size_t)(b * N_HEADS + head) * N_CH) * N_HW + m;
    const float* qkb = qk + ((size_t)(b * N_HEADS + head) * N_CH) * N_HW + m;
    float b_sq = 0.f;
    v16bf bsim;
#pragma unroll
    for (int c = 0; c < 16; ++c) {
        float e = qeb[c * N_HW];
        float k = qkb[c * N_HW];
        b_sq += e * k * k * k;
        if ((c >> 3) == hi) { bsim[c & 7] = (__bf16)(2.f * k * e); bsim[(c & 7) + 8] = (__bf16)e; }
    }

    const float* msb = ms + (size_t)b * N_THW;
    const float* mkb = mk + ((size_t)b * N_CK + head * N_CH) * N_THW;
    const float* mvb = mv + ((size_t)b * N_CV + head * N_CVH) * N_THW;

    v8f acc[4];
#pragma unroll
    for (int t = 0; t < 4; ++t) acc[t] = v8f{};

    const int cbase = hi * 8;
    const int nlane = q * 16 + ml;

    for (int it = 0; it < N_CHUNK; ++it) {
        const int n0 = it * 32;
        float msl = msb[n0 + nlane] * MS_SCALE;

        float mkv[8];
#pragma unroll
        for (int cc = 0; cc < 8; ++cc)
            mkv[cc] = mkb[(size_t)(cbase + cc) * N_THW + n0 + nlane];
        v16bf asim;
#pragma unroll
        for (int v = 0; v < 8; ++v)
#pragma unroll
            for (int j = 0; j < 2; ++j) {
                if (v < 4) {
                    asim[2 * v + j] = (__bf16)mkv[2 * v + j];
                } else {
                    float x = mkv[2 * v - 8 + j];
                    asim[2 * v + j] = (__bf16)(-x * x * x);
                }
            }

        v8f zero{};
        v8f sim = __builtin_amdgcn_wmma_f32_16x16x32_bf16(
            false, asim, false, bsim, (short)0, zero, false, false);

#pragma unroll
        for (int r = 0; r < 8; ++r) {
            float msr = __shfl(msl, r + 8 * hi, 32);
            float s = (sim[r] - b_sq) * msr;
            int nl = q * 16 + r + 8 * hi;
            expls[(head * 32 + nl) * 16 + ml] = EXP2F(s);
        }
        __syncthreads();
#pragma unroll
        for (int p = tid; p < 512; p += 256) {
            float d = expls[p] + expls[512 + p] + expls[1024 + p] + expls[1536 + p];
            rdls[p] = 1.0f / d;
        }
        __syncthreads();
        v16bf baff;
#pragma unroll
        for (int e = 0; e < 16; ++e) {
            int nl = 8 * hi + e + (e >= 8 ? 8 : 0);
            float a = expls[(head * 32 + nl) * 16 + ml] * rdls[nl * 16 + ml];
            baff[e] = (__bf16)a;
        }
#pragma unroll
        for (int t = 0; t < 4; ++t) {
            const float* row = mvb + (size_t)(q * 64 + t * 16 + ml) * N_THW + n0;
            v16bf amv;
#pragma unroll
            for (int v = 0; v < 8; ++v) {
                int kb = (v < 4) ? (cbase + 2 * v) : (16 + cbase + 2 * (v - 4));
                float2 mm = *(const float2*)(row + kb);
                amv[2 * v]     = (__bf16)mm.x;
                amv[2 * v + 1] = (__bf16)mm.y;
            }
            acc[t] = __builtin_amdgcn_wmma_f32_16x16x32_bf16(
                false, amv, false, baff, (short)0, acc[t], false, false);
        }
        __syncthreads();
    }

    float* ob = out + ((size_t)b * 2 * N_CV + head * N_CVH + q * 64) * N_HW + m0;
#pragma unroll
    for (int t = 0; t < 4; ++t)
#pragma unroll
        for (int r = 0; r < 8; ++r)
            ob[(size_t)(t * 16 + r + 8 * hi) * N_HW + ml] = acc[t][r];
}

// ---- output init: zero the mem channels (split-K target) + qv pass-through ----
__global__ __launch_bounds__(256)
void out_init_kernel(const float4* __restrict__ qv, float4* __restrict__ out)
{
    int idx  = blockIdx.x * 256 + threadIdx.x;     // 524288 float4 total
    int b    = idx >> 18;                          // 262144 float4 per batch
    int rest = idx & 262143;
    float4 v;
    if (rest < 131072) {
        v = float4{0.f, 0.f, 0.f, 0.f};            // mem channels [0, CV)
    } else {
        v = qv[(size_t)b * 131072 + (rest - 131072)];  // qv channels [CV, 2*CV)
    }
    out[idx] = v;
}

extern "C" void kernel_launch(void* const* d_in, const int* in_sizes, int n_in,
                              void* d_out, int out_size, void* d_ws, size_t ws_size,
                              hipStream_t stream)
{
    const float* mk = (const float*)d_in[0];
    const float* qk = (const float*)d_in[1];
    const float* ms = (const float*)d_in[2];
    const float* qe = (const float*)d_in[3];
    const float* mv = (const float*)d_in[4];
    const float* qv = (const float*)d_in[5];
    float* out = (float*)d_out;

    const size_t PMV_BYTES = (size_t)N_B * N_CV * N_THW * 2;          // 16 MB
    const size_t PMK_BYTES = (size_t)N_B * N_HEADS * N_THW * 32 * 2;  // 4 MB

    out_init_kernel<<<2048, 256, 0, stream>>>((const float4*)qv, (float4*)out);

    if (ws_size >= PMV_BYTES + PMK_BYTES) {
        __bf16* pmv = (__bf16*)d_ws;
        __bf16* pmk = (__bf16*)((char*)d_ws + PMV_BYTES);
        pack_mv_kernel<<<2048, 256, 0, stream>>>(mv, pmv);
        pack_mk_kernel<<<512, 256, 0, stream>>>(mk, pmk);
        mr_main_fast<<<N_SPLIT * N_B * (N_HW / 16), 256, 0, stream>>>(
            qk, ms, qe, pmk, pmv, out);
    } else {
        mr_main_kernel<<<N_B * (N_HW / 16), 256, 0, stream>>>(mk, qk, ms, qe, mv, out);
    }
}
